// GATVerifier_28690381537688
// MI455X (gfx1250) — matrix operand order repaired and checked by reading, not compiled, not run on hardware
//
#include <hip/hip_runtime.h>
#include <hip/hip_bf16.h>

typedef __attribute__((ext_vector_type(16))) _Float16 v16h;
typedef __attribute__((ext_vector_type(8)))  float    v8f;

// ---------- monotone float <-> uint key for atomicMax-based segment max ----------
__device__ __forceinline__ unsigned fkey(float x) {
    unsigned b = __float_as_uint(x);
    return (b & 0x80000000u) ? ~b : (b | 0x80000000u);
}
__device__ __forceinline__ float keyf(unsigned k) {
    return __uint_as_float((k & 0x80000000u) ? (k & 0x7FFFFFFFu) : ~k);
}

// ---------- pack W (64x64 f32, row-major, [K][Ncol]) into f16 WMMA B-fragment layout ----
// Packed index: ((((layer*3+mat)*4+nt)*2+kt)*32+lane)*16+s
// B element for lane,slot: K = kt*32 + (s<8 ? (lane>>4)*8 + s : 16 + (lane>>4)*8 + s-8) ; col = nt*16 + (lane&15)
__global__ void pack_weights(const float* __restrict__ Wl, const float* __restrict__ Wr,
                             const float* __restrict__ Wres, _Float16* __restrict__ Bpack) {
    int idx = blockIdx.x * blockDim.x + threadIdx.x;
    if (idx >= 3 * 3 * 4096) return;
    int s    =  idx        & 15;
    int lane = (idx >> 4)  & 31;
    int kt   = (idx >> 9)  &  1;
    int nt   = (idx >> 10) &  3;
    int mat  = (idx / 4096) % 3;
    int layer = idx / 12288;
    int K   = kt * 32 + ((s < 8) ? (lane >> 4) * 8 + s : 16 + (lane >> 4) * 8 + (s - 8));
    int col = nt * 16 + (lane & 15);
    const float* W = (mat == 0) ? Wl : (mat == 1) ? Wr : Wres;
    Bpack[idx] = (_Float16)W[layer * 4096 + K * 64 + col];
}

// ---------- encoder: h = concat(relu(x@Wt+bt), Et[type]) @ Wp + bp ; h0 = h ----------
__global__ __launch_bounds__(256)
void encoder(const float* __restrict__ x, const int* __restrict__ node_type,
             const float* __restrict__ Wt, const float* __restrict__ bt,
             const float* __restrict__ Et, const float* __restrict__ Wp,
             const float* __restrict__ bp, float* __restrict__ h,
             float* __restrict__ h0, int N) {
    __shared__ float buf[8][80];
    int wave = threadIdx.x >> 5, lane = threadIdx.x & 31;
    int n = blockIdx.x * 8 + wave;
    bool act = (n < N);
    int nn = act ? n : 0;
    float xv[8];
    const float* xp = x + (size_t)nn * 8;
#pragma unroll
    for (int f = 0; f < 8; ++f) xv[f] = xp[f];
#pragma unroll
    for (int j = 0; j < 2; ++j) {
        int d = lane + 32 * j;
        float a = bt[d];
#pragma unroll
        for (int f = 0; f < 8; ++f) a += xv[f] * Wt[f * 64 + d];
        buf[wave][d] = a > 0.f ? a : 0.f;
    }
    if (lane < 16) buf[wave][64 + lane] = Et[node_type[nn] * 16 + lane];
    __syncthreads();
#pragma unroll
    for (int j = 0; j < 2; ++j) {
        int d = lane + 32 * j;
        float a = bp[d];
        for (int k = 0; k < 80; ++k) a += buf[wave][k] * Wp[k * 64 + d];
        if (act) { h[(size_t)n * 64 + d] = a; h0[(size_t)n * 64 + d] = a; }
    }
}

// ---------- per-layer scratch init ----------
__global__ void layer_init(unsigned* __restrict__ mkey, float* __restrict__ ssum,
                           float* __restrict__ acc, int N) {
    int i = blockIdx.x * blockDim.x + threadIdx.x;
    if (i < N * 4) { mkey[i] = 0u; ssum[i] = 0.f; }
    if (i < N * 64) acc[i] = 0.f;
}

// ---------- WMMA GEMM: xl = h@Wl+bl, xr = h@Wr+br, res = h@Wres (wave per 16-node tile) --
__global__ __launch_bounds__(256)
void node_gemm3(const float* __restrict__ h, const _Float16* __restrict__ Bp,
                const float* __restrict__ bl, const float* __restrict__ br,
                float* __restrict__ xl, float* __restrict__ xr,
                float* __restrict__ res, int N) {
    int wave = threadIdx.x >> 5, lane = threadIdx.x & 31;
    int tile = blockIdx.x * 8 + wave;
    int ntiles = (N + 15) >> 4;
    if (tile >= ntiles) return;            // wave-uniform: EXEC stays all-ones for WMMA
    int m0 = tile << 4;
    bool full = (m0 + 16 <= N);            // wave-uniform fast-path condition
    int l = lane & 15, hi = lane >> 4;
    int row = m0 + l; if (row >= N) row = N - 1;
    const float* rp = h + (size_t)row * 64;

    // A fragments (16x32 f16): lane holds row l; slots 0..7 -> K=hi*8+s, 8..15 -> K=16+hi*8+s
    v16h a0, a1;
#pragma unroll
    for (int s = 0; s < 8; ++s) {
        a0[s]     = (_Float16)rp[hi * 8 + s];
        a0[s + 8] = (_Float16)rp[16 + hi * 8 + s];
        a1[s]     = (_Float16)rp[32 + hi * 8 + s];
        a1[s + 8] = (_Float16)rp[48 + hi * 8 + s];
    }

    float* outs[3] = {xl, xr, res};
    const float* biases[3] = {bl, br, nullptr};
#pragma unroll
    for (int mat = 0; mat < 3; ++mat) {
#pragma unroll
        for (int nt = 0; nt < 4; ++nt) {
            const _Float16* bp0 = Bp + ((size_t)((mat * 4 + nt) * 2) * 32 + lane) * 16;
            v16h b0 = *(const v16h*)bp0;
            v16h b1 = *(const v16h*)(bp0 + 512);   // kt=1 fragment
            v8f c = {};
            c = __builtin_amdgcn_wmma_f32_16x16x32_f16(false, a0, false, b0,
                                                       (short)0, c, false, false);
            c = __builtin_amdgcn_wmma_f32_16x16x32_f16(false, a1, false, b1,
                                                       (short)0, c, false, false);
            float bias = biases[mat] ? biases[mat][nt * 16 + l] : 0.f;
            float* op = outs[mat] + (size_t)m0 * 64 + nt * 16 + l;
            if (full) {
                // common case: unguarded stores (no per-element exec juggling)
#pragma unroll
                for (int v = 0; v < 8; ++v)        // D layout: M = v + 8*hi, Ncol = l
                    op[(size_t)(v + 8 * hi) * 64] = c[v] + bias;
            } else {
#pragma unroll
                for (int v = 0; v < 8; ++v) {
                    int r = v + 8 * hi;
                    if (m0 + r < N) op[(size_t)r * 64] = c[v] + bias;
                }
            }
        }
    }
}

// ---------- edge pass 1: scores + segment max via keyed atomicMax ----------
__global__ void edge_score(const float* __restrict__ xl, const float* __restrict__ xr,
                           const int* __restrict__ esrc, const int* __restrict__ edst,
                           const float* __restrict__ att, float* __restrict__ score,
                           unsigned* __restrict__ mkey, int E, int N) {
    int e = blockIdx.x * blockDim.x + threadIdx.x;
    int Etot = E + N;
    if (e >= Etot) return;
    int s = (e < E) ? esrc[e] : (e - E);
    int d = (e < E) ? edst[e] : (e - E);
    const float* pl = xl + (size_t)s * 64;
    const float* pr = xr + (size_t)d * 64;
    __builtin_prefetch(pl, 0, 1);   // xl[src] row is re-read by edge_accum
#pragma unroll
    for (int hh = 0; hh < 4; ++hh) {
        float sc = 0.f;
#pragma unroll
        for (int c = 0; c < 16; ++c) {
            float v = pl[hh * 16 + c] + pr[hh * 16 + c];
            v = v > 0.f ? v : 0.2f * v;            // leaky_relu 0.2
            sc += v * att[hh * 16 + c];
        }
        score[(size_t)e * 4 + hh] = sc;
        atomicMax(&mkey[(size_t)d * 4 + hh], fkey(sc));
    }
}

// ---------- edge pass 2: ex = exp(score-m); s += ex; acc += ex * xl[src] ----------
__global__ void edge_accum(const float* __restrict__ xl, const int* __restrict__ esrc,
                           const int* __restrict__ edst, const float* __restrict__ score,
                           const unsigned* __restrict__ mkey, float* __restrict__ ssum,
                           float* __restrict__ acc, int E, int N) {
    int e = blockIdx.x * blockDim.x + threadIdx.x;
    int Etot = E + N;
    if (e >= Etot) return;
    int s = (e < E) ? esrc[e] : (e - E);
    int d = (e < E) ? edst[e] : (e - E);
    const float* pl = xl + (size_t)s * 64;
    float* pa = acc + (size_t)d * 64;
#pragma unroll
    for (int hh = 0; hh < 4; ++hh) {
        float m  = keyf(mkey[(size_t)d * 4 + hh]);
        float ex = __expf(score[(size_t)e * 4 + hh] - m);
        atomicAdd(&ssum[(size_t)d * 4 + hh], ex);
#pragma unroll
        for (int c = 0; c < 16; ++c)
            atomicAdd(&pa[hh * 16 + c], ex * pl[hh * 16 + c]);
    }
}

// ---------- node update: normalize, +res +bconv, alpha-mix with h0, layernorm ----------
__global__ __launch_bounds__(256)
void node_update(const float* __restrict__ acc, const float* __restrict__ ssum,
                 const float* __restrict__ res, const float* __restrict__ bconv,
                 const float* __restrict__ h0, const float* __restrict__ gamma,
                 const float* __restrict__ beta, const float* __restrict__ alpha_p,
                 float* __restrict__ h, int N) {
    int wave = threadIdx.x >> 5, lane = threadIdx.x & 31;
    int n = blockIdx.x * 8 + wave;
    bool act = (n < N);
    int nn = act ? n : 0;
    float alpha = alpha_p[0];
    float y[2];
#pragma unroll
    for (int j = 0; j < 2; ++j) {
        int d = lane + 32 * j, hh = d >> 4;
        float denom = ssum[(size_t)nn * 4 + hh] + 1e-16f;
        float o = acc[(size_t)nn * 64 + d] / denom + res[(size_t)nn * 64 + d] + bconv[d];
        y[j] = alpha * o + (1.f - alpha) * h0[(size_t)nn * 64 + d];
    }
    float sum = y[0] + y[1];
    for (int off = 16; off; off >>= 1) sum += __shfl_xor(sum, off, 32);
    float mu = sum * (1.f / 64.f);
    float v0 = y[0] - mu, v1 = y[1] - mu;
    float vs = v0 * v0 + v1 * v1;
    for (int off = 16; off; off >>= 1) vs += __shfl_xor(vs, off, 32);
    float inv = rsqrtf(vs * (1.f / 64.f) + 1e-5f);
    if (act) {
#pragma unroll
        for (int j = 0; j < 2; ++j) {
            int d = lane + 32 * j;
            h[(size_t)n * 64 + d] = gamma[d] * (y[j] - mu) * inv + beta[d];
        }
    }
}

// ---------- head: sigmoid(relu(h@W1+b1)@W2+b2), wave per node ----------
__global__ __launch_bounds__(256)
void head(const float* __restrict__ h, const float* __restrict__ W1,
          const float* __restrict__ b1, const float* __restrict__ W2,
          const float* __restrict__ b2, float* __restrict__ out, int N) {
    int wave = threadIdx.x >> 5, lane = threadIdx.x & 31;
    int n = blockIdx.x * 8 + wave;
    bool act = (n < N);
    int nn = act ? n : 0;
    const float* hp = h + (size_t)nn * 64;
    float z = b1[lane];
    for (int k = 0; k < 64; ++k) z += hp[k] * W1[k * 32 + lane];
    z = z > 0.f ? z : 0.f;
    float r = z * W2[lane];
    for (int off = 16; off; off >>= 1) r += __shfl_xor(r, off, 32);
    if (act && lane == 0) out[n] = 1.f / (1.f + __expf(-(r + b2[0])));
}

extern "C" void kernel_launch(void* const* d_in, const int* in_sizes, int n_in,
                              void* d_out, int out_size, void* d_ws, size_t ws_size,
                              hipStream_t stream) {
    const float* x     = (const float*)d_in[0];
    const int*   ntype = (const int*)d_in[1];
    const int*   esrc  = (const int*)d_in[2];
    const int*   edst  = (const int*)d_in[3];
    const float* Wt    = (const float*)d_in[4];
    const float* bt    = (const float*)d_in[5];
    const float* Et    = (const float*)d_in[6];
    const float* Wp    = (const float*)d_in[7];
    const float* bp    = (const float*)d_in[8];
    const float* Wl    = (const float*)d_in[9];
    const float* bl    = (const float*)d_in[10];
    const float* Wr    = (const float*)d_in[11];
    const float* br    = (const float*)d_in[12];
    const float* att   = (const float*)d_in[13];
    const float* Wres  = (const float*)d_in[14];
    const float* bconv = (const float*)d_in[15];
    const float* gamma = (const float*)d_in[16];
    const float* beta  = (const float*)d_in[17];
    const float* alpha = (const float*)d_in[18];
    const float* W1    = (const float*)d_in[19];
    const float* b1    = (const float*)d_in[20];
    const float* W2    = (const float*)d_in[21];
    const float* b2    = (const float*)d_in[22];

    const int N = in_sizes[1];
    const int E = in_sizes[2];
    const int Etot = E + N;

    // workspace carve-out (256B-aligned slices)
    char* p = (char*)d_ws;
    auto take = [&](size_t bytes) { char* q = p; p += (bytes + 255) & ~(size_t)255; return q; };
    float*     h     = (float*)take((size_t)N * 64 * sizeof(float));
    float*     h0    = (float*)take((size_t)N * 64 * sizeof(float));
    float*     xl    = (float*)take((size_t)N * 64 * sizeof(float));
    float*     xr    = (float*)take((size_t)N * 64 * sizeof(float));
    float*     res   = (float*)take((size_t)N * 64 * sizeof(float));
    float*     acc   = (float*)take((size_t)N * 64 * sizeof(float));
    float*     ssum  = (float*)take((size_t)N * 4 * sizeof(float));
    unsigned*  mkey  = (unsigned*)take((size_t)N * 4 * sizeof(unsigned));
    float*     score = (float*)take((size_t)Etot * 4 * sizeof(float));
    _Float16*  Bpack = (_Float16*)take((size_t)3 * 3 * 4096 * sizeof(_Float16));

    const int ntiles = (N + 15) >> 4;

    pack_weights<<<(3 * 3 * 4096 + 255) / 256, 256, 0, stream>>>(Wl, Wr, Wres, Bpack);
    encoder<<<(N + 7) / 8, 256, 0, stream>>>(x, ntype, Wt, bt, Et, Wp, bp, h, h0, N);

    for (int l = 0; l < 3; ++l) {
        layer_init<<<(N * 64 + 255) / 256, 256, 0, stream>>>(mkey, ssum, acc, N);
        node_gemm3<<<(ntiles + 7) / 8, 256, 0, stream>>>(
            h, Bpack + (size_t)l * 12288, bl + l * 64, br + l * 64, xl, xr, res, N);
        edge_score<<<(Etot + 255) / 256, 256, 0, stream>>>(
            xl, xr, esrc, edst, att + l * 64, score, mkey, E, N);
        edge_accum<<<(Etot + 255) / 256, 256, 0, stream>>>(
            xl, esrc, edst, score, mkey, ssum, acc, E, N);
        node_update<<<(N + 7) / 8, 256, 0, stream>>>(
            acc, ssum, res, bconv + l * 64, h0, gamma + l * 64, beta + l * 64, alpha, h, N);
    }

    head<<<(N + 7) / 8, 256, 0, stream>>>(h, W1, b1, W2, b2, (float*)d_out, N);
}
